// SparseByteTransformer_79259326481045
// MI455X (gfx1250) — compile-verified, matrix operand order checked by reading, not run on hardware
//
#include <hip/hip_runtime.h>

// ---------------------------------------------------------------------------
// Problem dims (hardcoded from the reference)
// ---------------------------------------------------------------------------
#define BB 4
#define SS 2048
#define DD 512
#define HH 8
#define DHD 64
#define FFD 2048
#define LL 4
#define VV 256
#define MM (BB * SS)   // 8192 rows

typedef __attribute__((ext_vector_type(16))) __bf16 v16bf;
typedef __attribute__((ext_vector_type(8)))  float  v8f;
typedef __attribute__((ext_vector_type(4)))  unsigned int v4u;
typedef __attribute__((ext_vector_type(8)))  int v8i_;
typedef __attribute__((ext_vector_type(4)))  int v4i_;

// ---------------------------------------------------------------------------
// bf16 helpers (storage = unsigned short bit pattern)
// ---------------------------------------------------------------------------
__device__ __forceinline__ unsigned short f2bf(float f) {
    unsigned int u = __float_as_uint(f);
    unsigned int r = 0x7FFFu + ((u >> 16) & 1u);
    u += r;
    return (unsigned short)(u >> 16);
}

__device__ __forceinline__ __bf16 bits2bf(unsigned short u) {
    union { unsigned short s; __bf16 b; } x;
    x.s = u;
    return x.b;
}

__device__ __forceinline__ v8f zero_v8f() {
    v8f z;
#pragma unroll
    for (int i = 0; i < 8; ++i) z[i] = 0.0f;
    return z;
}

// ---------------------------------------------------------------------------
// Tensor Data Mover: 2-D bf16 tile (tile0 x tile1, row stride strideElems)
// global -> LDS, with LDS padding of 4 DWORDs (8 bf16) every 16 DWORDs
// (32 bf16) => padded LDS row stride of 40 ushorts.
// D# packing per CDNA5 ISA.
// ---------------------------------------------------------------------------
__device__ __forceinline__ void tdm_load_2d(unsigned ldsOff, const unsigned short* g,
                                            unsigned tile0, unsigned tile1,
                                            unsigned long long strideElems) {
    unsigned long long ga = (unsigned long long)(uintptr_t)g;
    v4u g0;
    g0[0] = 1u;                                        // count=1, user mode
    g0[1] = ldsOff;                                    // lds_addr (bytes)
    g0[2] = (unsigned)(ga & 0xFFFFFFFFu);              // global_addr[31:0]
    g0[3] = (unsigned)((ga >> 32) & 0x01FFFFFFu)       // global_addr[56:32]
          | (2u << 30);                                // type=2 ("image")
    v8i_ g1;
    g1[0] = (int)((1u << 16)                           // data_size = 2 bytes
          | (1u << 20)                                 // pad_enable
          | (3u << 22)                                 // pad_interval: 16 DWORDs
          | (3u << 25));                               // pad_amount: 4 DWORDs
    g1[1] = (int)((tile0 & 0xFFFFu) << 16);            // tensor_dim0[15:0]
    g1[2] = (int)((tile1 & 0xFFFFu) << 16);            // tensor_dim0 hi=0 | tensor_dim1 lo
    g1[3] = (int)((tile0 & 0xFFFFu) << 16);            // tensor_dim1 hi=0 | tile_dim0
    g1[4] = (int)(tile1 & 0xFFFFu);                    // tile_dim1 | tile_dim2=0
    g1[5] = (int)(unsigned)(strideElems & 0xFFFFFFFFull);        // dim0_stride lo
    g1[6] = (int)(unsigned)((strideElems >> 32) & 0xFFFFull);    // dim0_stride hi | dim1_stride lo
    g1[7] = 0;                                         // dim1_stride hi
    v4i_ z4 = {0, 0, 0, 0};
#if defined(__clang_major__) && (__clang_major__ >= 23)
    v8i_ z8 = {0, 0, 0, 0, 0, 0, 0, 0};
    __builtin_amdgcn_tensor_load_to_lds(g0, g1, z4, z4, z8, 0);
#else
    __builtin_amdgcn_tensor_load_to_lds(g0, g1, z4, z4, 0);
#endif
}

// ---------------------------------------------------------------------------
// Fragment loaders (CDNA5 wave32 WMMA operand layout)
// ---------------------------------------------------------------------------
__device__ __forceinline__ v16bf load_frag_lds(const unsigned short* base,
                                               int stride, int r16, int lane) {
    int row = r16 + (lane & 15);
    int bk  = (lane >> 4) << 3;
    const unsigned short* p = base + row * stride + bk;
    v16bf a;
#pragma unroll
    for (int i = 0; i < 8; ++i) a[i] = bits2bf(p[i]);
#pragma unroll
    for (int i = 0; i < 8; ++i) a[8 + i] = bits2bf(p[16 + i]);
    return a;
}

__device__ __forceinline__ v16bf load_frag_g(const unsigned short* base,
                                             int ld, int kbase, int lane) {
    int row = lane & 15;
    int bk  = kbase + ((lane >> 4) << 3);
    const unsigned short* p = base + (size_t)row * ld + bk;
    v16bf a;
#pragma unroll
    for (int i = 0; i < 8; ++i) a[i] = bits2bf(p[i]);
#pragma unroll
    for (int i = 0; i < 8; ++i) a[8 + i] = bits2bf(p[16 + i]);
    return a;
}

__device__ __forceinline__ v16bf load_frag_v(const unsigned short* vbase, int lane) {
    int col = lane & 15;
    int bk  = (lane >> 4) << 3;
    const unsigned short* p = vbase + col + (size_t)bk * DD;
    v16bf b;
#pragma unroll
    for (int i = 0; i < 8; ++i) b[i] = bits2bf(p[(size_t)i * DD]);
#pragma unroll
    for (int i = 0; i < 8; ++i) b[8 + i] = bits2bf(p[(size_t)(16 + i) * DD]);
    return b;
}

__device__ __forceinline__ float red16_max(float v) {
#pragma unroll
    for (int m = 1; m < 16; m <<= 1) v = fmaxf(v, __shfl_xor(v, m, 32));
    return v;
}
__device__ __forceinline__ float red16_sum(float v) {
#pragma unroll
    for (int m = 1; m < 16; m <<= 1) v += __shfl_xor(v, m, 32);
    return v;
}

// ---------------------------------------------------------------------------
// fp32 -> bf16 conversion (grid-stride)
// ---------------------------------------------------------------------------
__global__ void f32_to_bf16_kernel(const float* __restrict__ in,
                                   unsigned short* __restrict__ out, int n) {
    int i = blockIdx.x * blockDim.x + threadIdx.x;
    int stride = gridDim.x * blockDim.x;
    for (; i < n; i += stride) out[i] = f2bf(in[i]);
}

// [L,K,N] fp32 -> per-layer transposed [L][N][K] bf16
__global__ void convT_kernel(const float* __restrict__ in,
                             unsigned short* __restrict__ out,
                             int Kd, int Nd, int total) {
    int i = blockIdx.x * blockDim.x + threadIdx.x;
    int stride = gridDim.x * blockDim.x;
    int KN = Kd * Nd;
    for (; i < total; i += stride) {
        int l = i / KN;
        int r = i - l * KN;
        int k = r / Nd;
        int n = r - k * Nd;
        out[(size_t)l * KN + (size_t)n * Kd + k] = f2bf(in[i]);
    }
}

// ---------------------------------------------------------------------------
// Embedding + sinusoidal positional encoding
// ---------------------------------------------------------------------------
__global__ __launch_bounds__(256) void embed_kernel(const int* __restrict__ tok,
                                                    const float* __restrict__ emb,
                                                    float* __restrict__ xf,
                                                    unsigned short* __restrict__ xb) {
    int rs = blockIdx.x;        // 0 .. B*S-1
    int s  = rs & (SS - 1);
    int t  = tok[rs];
    const float sq   = 22.6274169979695f;   // sqrt(512)
    const float cexp = -0.0359778530749f;   // -2*ln(10000)/512
    for (int d = threadIdx.x; d < DD; d += 256) {
        int   pr  = d >> 1;
        float ang = (float)s * __expf(cexp * (float)pr);
        float pe  = (d & 1) ? __cosf(ang) : __sinf(ang);
        float val = emb[(size_t)t * DD + d] * sq + pe;
        size_t off = (size_t)rs * DD + d;
        xf[off] = val;
        xb[off] = f2bf(val);
    }
}

// ---------------------------------------------------------------------------
// Tiled bf16 WMMA GEMM with TDM double-buffered tile staging:
//   C[M,N] = A[M,K] @ Bt[N,K]^T + bias (optional ReLU)
// Block tile 128x64, BK=32, 256 threads = 8 waves (4x2), each wave 32x32.
// Both operands are K-contiguous, fetched by tensor_load_to_lds with padding
// (padded LDS row stride = 40 ushorts).
// ---------------------------------------------------------------------------
#define BM 128
#define BN 64
#define BK 32
#define LDS_STRIDE 40

__global__ __launch_bounds__(256) void gemm_bf16_kernel(
    const unsigned short* __restrict__ A, const unsigned short* __restrict__ Bt,
    const float* __restrict__ bias, float* __restrict__ Cf,
    unsigned short* __restrict__ Cb, int M, int N, int K, int relu) {
    __shared__ unsigned short As[2][BM * LDS_STRIDE];
    __shared__ unsigned short Bs[2][BN * LDS_STRIDE];

    int t    = threadIdx.x;
    int lane = t & 31;
    int w    = t >> 5;
    int mw   = w >> 1;  // 0..3
    int nw   = w & 1;   // 0..1
    int m0   = blockIdx.y * BM;
    int n0   = blockIdx.x * BN;

    v8f acc[2][2];
#pragma unroll
    for (int im = 0; im < 2; ++im)
#pragma unroll
        for (int in = 0; in < 2; ++in) acc[im][in] = zero_v8f();

    int ktiles = K / BK;

    // prologue: DMA tile 0 (wave 0 drives the TDM)
    if (w == 0) {
        tdm_load_2d((unsigned)(uintptr_t)&As[0][0],
                    A + (size_t)m0 * K, BK, BM, (unsigned long long)K);
        tdm_load_2d((unsigned)(uintptr_t)&Bs[0][0],
                    Bt + (size_t)n0 * K, BK, BN, (unsigned long long)K);
    }

    for (int kt = 0; kt < ktiles; ++kt) {
        int buf = kt & 1;
        if (w == 0) {
            if (kt + 1 < ktiles) {
                int nb = buf ^ 1;
                int k1 = (kt + 1) * BK;
                tdm_load_2d((unsigned)(uintptr_t)&As[nb][0],
                            A + (size_t)m0 * K + k1, BK, BM, (unsigned long long)K);
                tdm_load_2d((unsigned)(uintptr_t)&Bs[nb][0],
                            Bt + (size_t)n0 * K + k1, BK, BN, (unsigned long long)K);
                __builtin_amdgcn_s_wait_tensorcnt(2);   // tile kt landed
            } else {
                __builtin_amdgcn_s_wait_tensorcnt(0);   // last tile landed
            }
        }
        __syncthreads();
#pragma unroll
        for (int im = 0; im < 2; ++im) {
            v16bf af = load_frag_lds(&As[buf][0], LDS_STRIDE, mw * 32 + im * 16, lane);
#pragma unroll
            for (int in = 0; in < 2; ++in) {
                v16bf bf = load_frag_lds(&Bs[buf][0], LDS_STRIDE, nw * 32 + in * 16, lane);
                acc[im][in] = __builtin_amdgcn_wmma_f32_16x16x32_bf16(
                    false, af, false, bf, (short)0, acc[im][in], false, false);
            }
        }
        __syncthreads();   // all waves done with buf before it is refilled
    }

    // epilogue: bias (+ReLU) then fp32 and/or bf16 stores
    int rowhalf = lane >> 4;
    int cn      = lane & 15;
#pragma unroll
    for (int im = 0; im < 2; ++im) {
#pragma unroll
        for (int in = 0; in < 2; ++in) {
            int gmBase = m0 + mw * 32 + im * 16 + rowhalf * 8;
            int gn     = n0 + nw * 32 + in * 16 + cn;
            float bv   = bias ? bias[gn] : 0.0f;
#pragma unroll
            for (int r = 0; r < 8; ++r) {
                float v = acc[im][in][r] + bv;
                if (relu) v = fmaxf(v, 0.0f);
                size_t off = (size_t)(gmBase + r) * N + gn;
                if (Cf) Cf[off] = v;
                if (Cb) Cb[off] = f2bf(v);
            }
        }
    }
}

// ---------------------------------------------------------------------------
// Sparse flash attention: one wave per (16-query block, head, batch).
// scores = Q K^T / 8 with mask: causal & (local<256 | (i-j)%128==0 | j<16)
// ---------------------------------------------------------------------------
__device__ __forceinline__ float mask_score(float s, int i, int j) {
    int d = i - j;
    bool allow = (d >= 0) && ((d < 256) || ((d & 127) == 0) || (j < 16));
    return allow ? s : -1e9f;
}

__global__ __launch_bounds__(32) void attn_kernel(const unsigned short* __restrict__ Q,
                                                  const unsigned short* __restrict__ Kt,
                                                  const unsigned short* __restrict__ Vt,
                                                  unsigned short* __restrict__ O) {
    __shared__ unsigned short p_lds[16 * 40];
    int lane = threadIdx.x;
    int q0   = blockIdx.x * 16;
    int h    = blockIdx.y;
    int b    = blockIdx.z;

    const unsigned short* qbase = Q + ((size_t)(b * SS + q0)) * DD + h * DHD;
    v16bf qf0 = load_frag_g(qbase, DD, 0, lane);
    v16bf qf1 = load_frag_g(qbase, DD, 32, lane);

    int rowhalf = lane >> 4;
    int cn      = lane & 15;

    float m[8], ssum[8];
    v8f oacc[4];
#pragma unroll
    for (int r = 0; r < 8; ++r) { m[r] = -1e30f; ssum[r] = 0.0f; }
#pragma unroll
    for (int c = 0; c < 4; ++c) oacc[c] = zero_v8f();

    const float scale = 0.125f;  // 1/sqrt(64)
    int kbmax = (q0 + 15) >> 5;
    for (int kb = 0; kb <= kbmax; ++kb) {
        int j0 = kb * 32;
        // relevance: global | local | strided
        bool rel = (j0 < 16);
        rel = rel || (j0 + 31 >= q0 - 255);
        {
            int ri = q0 & 127, rj = j0 & 127;
            rel = rel || ((ri < rj + 32) && (rj < ri + 16));
        }
        if (!rel) continue;

        // ---- scores (two 16x16 tiles, K-dim 64 = 2 WMMA steps)
        const unsigned short* kbase0 = Kt + ((size_t)(b * SS + j0)) * DD + h * DHD;
        const unsigned short* kbase1 = kbase0 + (size_t)16 * DD;
        v8f s0 = __builtin_amdgcn_wmma_f32_16x16x32_bf16(
            false, qf0, false, load_frag_g(kbase0, DD, 0, lane), (short)0, zero_v8f(), false, false);
        s0 = __builtin_amdgcn_wmma_f32_16x16x32_bf16(
            false, qf1, false, load_frag_g(kbase0, DD, 32, lane), (short)0, s0, false, false);
        v8f s1 = __builtin_amdgcn_wmma_f32_16x16x32_bf16(
            false, qf0, false, load_frag_g(kbase1, DD, 0, lane), (short)0, zero_v8f(), false, false);
        s1 = __builtin_amdgcn_wmma_f32_16x16x32_bf16(
            false, qf1, false, load_frag_g(kbase1, DD, 32, lane), (short)0, s1, false, false);

        // ---- scale + mask, per-row block max (16-lane half reductions)
        float bm[8];
#pragma unroll
        for (int r = 0; r < 8; ++r) {
            int i = q0 + r + rowhalf * 8;
            float a0 = mask_score(s0[r] * scale, i, j0 + cn);
            float a1 = mask_score(s1[r] * scale, i, j0 + 16 + cn);
            s0[r] = a0;
            s1[r] = a1;
            bm[r] = red16_max(fmaxf(a0, a1));
        }
        float fac[8];
#pragma unroll
        for (int r = 0; r < 8; ++r) {
            float nm = fmaxf(m[r], bm[r]);
            fac[r]   = __expf(m[r] - nm);
            m[r]     = nm;
        }
        // ---- p = exp(s - m), row sums, update running sum
#pragma unroll
        for (int r = 0; r < 8; ++r) {
            float p0 = __expf(s0[r] - m[r]);
            float p1 = __expf(s1[r] - m[r]);
            s0[r] = p0;
            s1[r] = p1;
            float rs = red16_sum(p0 + p1);
            ssum[r]  = ssum[r] * fac[r] + rs;
        }
        // ---- rescale O accumulators
#pragma unroll
        for (int c = 0; c < 4; ++c)
#pragma unroll
            for (int r = 0; r < 8; ++r) oacc[c][r] *= fac[r];

        // ---- P tile through LDS (C layout -> A-fragment layout)
        __syncthreads();
#pragma unroll
        for (int r = 0; r < 8; ++r) {
            p_lds[(r + rowhalf * 8) * 40 + cn]      = f2bf(s0[r]);
            p_lds[(r + rowhalf * 8) * 40 + 16 + cn] = f2bf(s1[r]);
        }
        __syncthreads();
        v16bf pf = load_frag_lds(p_lds, 40, 0, lane);

        // ---- O += P @ V (dh = 64 -> 4 column chunks)
        const unsigned short* vbase = Vt + ((size_t)(b * SS + j0)) * DD + h * DHD;
#pragma unroll
        for (int c = 0; c < 4; ++c) {
            v16bf vf = load_frag_v(vbase + c * 16, lane);
            oacc[c] = __builtin_amdgcn_wmma_f32_16x16x32_bf16(
                false, pf, false, vf, (short)0, oacc[c], false, false);
        }
    }

    // ---- normalize and store (bf16 for the Wo GEMM)
    unsigned short* obase = O + ((size_t)(b * SS + q0)) * DD + h * DHD;
#pragma unroll
    for (int r = 0; r < 8; ++r) {
        float inv = 1.0f / ssum[r];
#pragma unroll
        for (int c = 0; c < 4; ++c) {
            obase[(size_t)(r + rowhalf * 8) * DD + c * 16 + cn] = f2bf(oacc[c][r] * inv);
        }
    }
}

// ---------------------------------------------------------------------------
// Residual add + LayerNorm over D=512 (one block per row, 256 threads)
// ---------------------------------------------------------------------------
__global__ __launch_bounds__(256) void add_ln_kernel(float* __restrict__ x,
                                                     const float* __restrict__ pj,
                                                     const float* __restrict__ g,
                                                     const float* __restrict__ bt,
                                                     unsigned short* __restrict__ xb) {
    __shared__ float red[8];
    int row = blockIdx.x;
    int t   = threadIdx.x;
    size_t base = (size_t)row * DD;
    float v0 = x[base + t] + pj[base + t];
    float v1 = x[base + t + 256] + pj[base + t + 256];
    float s  = v0 + v1;
#pragma unroll
    for (int m2 = 1; m2 < 32; m2 <<= 1) s += __shfl_xor(s, m2, 32);
    if ((t & 31) == 0) red[t >> 5] = s;
    __syncthreads();
    float tot = red[0] + red[1] + red[2] + red[3] + red[4] + red[5] + red[6] + red[7];
    float mean = tot * (1.0f / DD);
    float d0 = v0 - mean, d1 = v1 - mean;
    float sq = d0 * d0 + d1 * d1;
#pragma unroll
    for (int m2 = 1; m2 < 32; m2 <<= 1) sq += __shfl_xor(sq, m2, 32);
    __syncthreads();
    if ((t & 31) == 0) red[t >> 5] = sq;
    __syncthreads();
    float var  = (red[0] + red[1] + red[2] + red[3] + red[4] + red[5] + red[6] + red[7]) * (1.0f / DD);
    float rstd = rsqrtf(var + 1e-5f);
    float o0 = d0 * rstd * g[t] + bt[t];
    float o1 = d1 * rstd * g[t + 256] + bt[t + 256];
    x[base + t]        = o0;
    x[base + t + 256]  = o1;
    xb[base + t]       = f2bf(o0);
    xb[base + t + 256] = f2bf(o1);
}

// ---------------------------------------------------------------------------
// Host-side orchestration
// ---------------------------------------------------------------------------
extern "C" void kernel_launch(void* const* d_in, const int* in_sizes, int n_in,
                              void* d_out, int out_size, void* d_ws, size_t ws_size,
                              hipStream_t stream) {
    (void)in_sizes; (void)n_in; (void)out_size; (void)ws_size;

    const int*   tokens = (const int*)d_in[0];
    const float* emb    = (const float*)d_in[1];
    const float* b_out  = (const float*)d_in[2];
    const float* Wq     = (const float*)d_in[3];
    const float* bq     = (const float*)d_in[4];
    const float* Wk     = (const float*)d_in[5];
    const float* bk_p   = (const float*)d_in[6];
    const float* Wv     = (const float*)d_in[7];
    const float* bv     = (const float*)d_in[8];
    const float* Wo     = (const float*)d_in[9];
    const float* bo     = (const float*)d_in[10];
    const float* ln1g   = (const float*)d_in[11];
    const float* ln1b   = (const float*)d_in[12];
    const float* W1     = (const float*)d_in[13];
    const float* b1     = (const float*)d_in[14];
    const float* W2     = (const float*)d_in[15];
    const float* b2     = (const float*)d_in[16];
    const float* ln2g   = (const float*)d_in[17];
    const float* ln2b   = (const float*)d_in[18];

    char* p = (char*)d_ws;
    auto alloc = [&](size_t bytes) -> void* {
        void* r = (void*)p;
        p += (bytes + 255) & ~(size_t)255;
        return r;
    };

    float*          xf    = (float*)alloc((size_t)MM * DD * 4);
    unsigned short* xb    = (unsigned short*)alloc((size_t)MM * DD * 2);
    unsigned short* qb    = (unsigned short*)alloc((size_t)MM * DD * 2);
    unsigned short* kb    = (unsigned short*)alloc((size_t)MM * DD * 2);
    unsigned short* vb    = (unsigned short*)alloc((size_t)MM * DD * 2);
    unsigned short* ob    = (unsigned short*)alloc((size_t)MM * DD * 2);
    float*          proj  = (float*)alloc((size_t)MM * DD * 4);
    unsigned short* hb    = (unsigned short*)alloc((size_t)MM * FFD * 2);
    unsigned short* wq_b  = (unsigned short*)alloc((size_t)LL * DD * DD * 2);   // [l][N][K]
    unsigned short* wk_b  = (unsigned short*)alloc((size_t)LL * DD * DD * 2);
    unsigned short* wv_b  = (unsigned short*)alloc((size_t)LL * DD * DD * 2);
    unsigned short* wo_b  = (unsigned short*)alloc((size_t)LL * DD * DD * 2);
    unsigned short* w1_b  = (unsigned short*)alloc((size_t)LL * DD * FFD * 2);  // [l][FF][D]
    unsigned short* w2_b  = (unsigned short*)alloc((size_t)LL * FFD * DD * 2);  // [l][D][FF]
    unsigned short* emb_b = (unsigned short*)alloc((size_t)VV * DD * 2);        // [V][D] == Bt

    // ---- one-time weight conversion + transpose (B operands stored [N][K])
    const int nDD = LL * DD * DD;
    const int nDF = LL * DD * FFD;
    convT_kernel<<<1024, 256, 0, stream>>>(Wq, wq_b, DD, DD, nDD);
    convT_kernel<<<1024, 256, 0, stream>>>(Wk, wk_b, DD, DD, nDD);
    convT_kernel<<<1024, 256, 0, stream>>>(Wv, wv_b, DD, DD, nDD);
    convT_kernel<<<1024, 256, 0, stream>>>(Wo, wo_b, DD, DD, nDD);
    convT_kernel<<<1024, 256, 0, stream>>>(W1, w1_b, DD, FFD, nDF);
    convT_kernel<<<1024, 256, 0, stream>>>(W2, w2_b, FFD, DD, nDF);
    f32_to_bf16_kernel<<<1024, 256, 0, stream>>>(emb, emb_b, VV * DD);  // already [N=V][K=D]

    // ---- embedding + positional encoding
    embed_kernel<<<MM, 256, 0, stream>>>(tokens, emb, xf, xb);

    dim3 gD(DD / BN, MM / BM);    // N=512
    dim3 gF(FFD / BN, MM / BM);   // N=2048
    dim3 gV(VV / BN, MM / BM);    // N=256
    dim3 gA(SS / 16, HH, BB);

    for (int l = 0; l < LL; ++l) {
        const unsigned short* wq_l = wq_b + (size_t)l * DD * DD;
        const unsigned short* wk_l = wk_b + (size_t)l * DD * DD;
        const unsigned short* wv_l = wv_b + (size_t)l * DD * DD;
        const unsigned short* wo_l = wo_b + (size_t)l * DD * DD;
        const unsigned short* w1_l = w1_b + (size_t)l * DD * FFD;
        const unsigned short* w2_l = w2_b + (size_t)l * FFD * DD;

        // Q, K, V projections (bf16 out)
        gemm_bf16_kernel<<<gD, 256, 0, stream>>>(xb, wq_l, bq + l * DD, nullptr, qb, MM, DD, DD, 0);
        gemm_bf16_kernel<<<gD, 256, 0, stream>>>(xb, wk_l, bk_p + l * DD, nullptr, kb, MM, DD, DD, 0);
        gemm_bf16_kernel<<<gD, 256, 0, stream>>>(xb, wv_l, bv + l * DD, nullptr, vb, MM, DD, DD, 0);

        // sparse flash attention
        attn_kernel<<<gA, 32, 0, stream>>>(qb, kb, vb, ob);

        // output projection (fp32 out), residual + LN1
        gemm_bf16_kernel<<<gD, 256, 0, stream>>>(ob, wo_l, bo + l * DD, proj, nullptr, MM, DD, DD, 0);
        add_ln_kernel<<<MM, 256, 0, stream>>>(xf, proj, ln1g + l * DD, ln1b + l * DD, xb);

        // FFN: ReLU(x W1 + b1) W2 + b2, residual + LN2
        gemm_bf16_kernel<<<gF, 256, 0, stream>>>(xb, w1_l, b1 + l * FFD, nullptr, hb, MM, FFD, DD, 1);
        gemm_bf16_kernel<<<gD, 256, 0, stream>>>(hb, w2_l, b2 + l * DD, proj, nullptr, MM, DD, FFD, 0);
        add_ln_kernel<<<MM, 256, 0, stream>>>(xf, proj, ln2g + l * DD, ln2b + l * DD, xb);
    }

    // ---- tied output projection -> d_out [B,S,V] fp32
    gemm_bf16_kernel<<<gV, 256, 0, stream>>>(xb, emb_b, b_out, (float*)d_out, nullptr, MM, VV, DD, 0);
}
